// MobileBertAttention_58780922413790
// MI455X (gfx1250) — matrix operand order repaired and checked by reading, not compiled, not run on hardware
//
#include <hip/hip_runtime.h>

#define DEV_INLINE __device__ __forceinline__

typedef __attribute__((ext_vector_type(16))) _Float16 v16h;
typedef __attribute__((ext_vector_type(8)))  _Float16 v8h;
typedef __attribute__((ext_vector_type(2)))  _Float16 v2h;
typedef __attribute__((ext_vector_type(8)))  float    v8f;
typedef __attribute__((ext_vector_type(4)))  unsigned int v4u;
typedef __attribute__((ext_vector_type(8)))  int      v8i;
typedef __attribute__((ext_vector_type(4)))  int      v4i;

static constexpr int Bb  = 8;
static constexpr int Ss  = 2048;
static constexpr int HID = 512;
static constexpr int NH  = 8;
static constexpr int HD  = 64;

// ---------------------------------------------------------------------------
// WMMA fragment helpers (CDNA5 wave32, V_WMMA_F32_16X16X32_F16)
// A 16x32 f16: lane L holds row m=L%16; halfs h: K = 16*(h>>3) + 8*(L>>4) + (h&7)
//   -> two contiguous 8-half (16B) runs at +8g and +16+8g.
// B 32x16 f16: lane L holds col n=L%16; halfs h: K = 16*(h>>3) + 8*(L>>4) + (h&7)
//   -> two contiguous 8-half (16B) runs at +8g and +16+8g.
// C/D 16x16 f32: lane L holds col n=L%16; VGPR r holds row m = r + 8*(L>>4).
// ---------------------------------------------------------------------------
DEV_INLINE v16h frag_a(const _Float16* p, int ld) {
  const int lane = threadIdx.x & 31;
  const int m = lane & 15, g = lane >> 4;
  const _Float16* r = p + (size_t)m * ld + 8 * g;
  v8h lo = *(const v8h*)(r);
  v8h hi = *(const v8h*)(r + 16);
  return __builtin_shufflevector(lo, hi, 0,1,2,3,4,5,6,7,8,9,10,11,12,13,14,15);
}

DEV_INLINE v16h frag_b(const _Float16* p, int ld) {
  const int lane = threadIdx.x & 31;
  const int n = lane & 15, g = lane >> 4;
  const _Float16* r = p + (size_t)n * ld + 8 * g;
  v8h lo = *(const v8h*)(r);
  v8h hi = *(const v8h*)(r + 16);
  return __builtin_shufflevector(lo, hi, 0,1,2,3,4,5,6,7,8,9,10,11,12,13,14,15);
}

DEV_INLINE v8f wmma32(v16h a, v16h b, v8f c) {
  return __builtin_amdgcn_wmma_f32_16x16x32_f16(
      false, a, false, b, (short)0, c, false, false);
}

// ---------------------------------------------------------------------------
// Tensor Data Mover: DMA a 2D f16 tile (tile_d0 x tile_d1, row stride
// stride0 elements) from global memory into LDS at lds_addr.  Descriptor
// bitfields per CDNA5 ISA.  Tracked by
// TENSORcnt; caller must s_wait_tensorcnt + barrier before consuming.
// ---------------------------------------------------------------------------
DEV_INLINE void tdm_load_2d_f16(unsigned int lds_addr, const _Float16* gptr,
                                unsigned int tensor_d0, unsigned int tensor_d1,
                                unsigned int tile_d0, unsigned int tile_d1,
                                unsigned int stride0) {
  const unsigned long long ga = (unsigned long long)(uintptr_t)gptr;
  v4u g0;
  g0[0] = 1u;                                      // count=1, user mode
  g0[1] = lds_addr;                                // LDS byte address
  g0[2] = (unsigned int)ga;                        // global_addr[31:0]
  g0[3] = (unsigned int)(ga >> 32) | 0x80000000u;  // global_addr[56:32], type=2
  v8i g1;
  g1[0] = (int)(1u << 16);                         // data_size = 1 -> 2 bytes
  g1[1] = (int)((tensor_d0 & 0xffffu) << 16);              // tensor_dim0[15:0]
  g1[2] = (int)((tensor_d0 >> 16) | ((tensor_d1 & 0xffffu) << 16));
  g1[3] = (int)((tensor_d1 >> 16) | (tile_d0 << 16));      // tile_dim0
  g1[4] = (int)tile_d1;                            // tile_dim1 (tile_dim2 = 0)
  g1[5] = (int)stride0;                            // tensor_dim0_stride[31:0]
  g1[6] = 0;
  g1[7] = 0;
  v4i z4 = {};
#if defined(__clang_major__) && (__clang_major__ >= 23)
  v8i z8 = {};
  __builtin_amdgcn_tensor_load_to_lds(g0, g1, z4, z4, z8, 0);
#else
  __builtin_amdgcn_tensor_load_to_lds(g0, g1, z4, z4, 0);
#endif
}

// ---------------------------------------------------------------------------
// Kernel 0: f32 -> f16 convert (weights)
// ---------------------------------------------------------------------------
__global__ void cvt_f32_f16(const float* __restrict__ src,
                            _Float16* __restrict__ dst, int n) {
  int i = blockIdx.x * blockDim.x + threadIdx.x;
  if (i < n) dst[i] = (_Float16)src[i];
}

// ---------------------------------------------------------------------------
// Kernel 1: projection GEMM  Y = X @ W^T + bias  (f32 in, f16 out)
//   transpose_out==0: Y[((b*NH+h)*S + s)*HD + d]   (Q,K)
//   transpose_out==1: Y[((b*NH+h)*HD + d)*S + s]   (V transposed)
// grid.x = M/16 = 1024, block = 256 (8 waves; wave w owns N columns w*64..+63)
// ---------------------------------------------------------------------------
__global__ __launch_bounds__(256)
void proj_kernel(const float* __restrict__ X,
                 const _Float16* __restrict__ W16,
                 const float* __restrict__ bias,
                 _Float16* __restrict__ Y,
                 int transpose_out) {
  __shared__ _Float16 ldsA[16 * 32];

  const int m0   = blockIdx.x * 16;
  const int wave = threadIdx.x >> 5;
  const int lane = threadIdx.x & 31;
  const int nl   = lane & 15, g = lane >> 4;
  const int n_base = wave * 64;

  // static staging assignment: 512 elements / 256 threads = 2 contiguous each
  const int sr = threadIdx.x >> 4;          // row 0..15
  const int sc = (threadIdx.x & 15) * 2;    // col 0,2,..,30

  v8f acc[4] = {};

  for (int k0 = 0; k0 < HID; k0 += 32) {
    // stage X tile (16x32) f32 -> f16 into LDS: one b64 load + one b32 store
    {
      const float2 xv =
          *(const float2*)(X + (size_t)(m0 + sr) * HID + k0 + sc);
      v2h hv;
      hv[0] = (_Float16)xv.x;
      hv[1] = (_Float16)xv.y;
      *(v2h*)(ldsA + sr * 32 + sc) = hv;
    }
    __syncthreads();
    v16h a = frag_a(ldsA, 32);
#pragma unroll
    for (int t = 0; t < 4; ++t) {
      v16h b = frag_b(W16 + (size_t)(n_base + t * 16) * HID + k0, HID);
      acc[t] = wmma32(a, b, acc[t]);
    }
    __syncthreads();
  }

#pragma unroll
  for (int t = 0; t < 4; ++t) {
    const int n_glob = n_base + t * 16 + nl;
    const int h = n_glob >> 6;
    const int d = n_glob & 63;
    const float bval = bias[n_glob];
#pragma unroll
    for (int r = 0; r < 8; ++r) {
      const int m = m0 + r + 8 * g;
      const int bb = m / Ss, s = m % Ss;
      const float val = acc[t][r] + bval;
      size_t off = transpose_out
          ? (((size_t)(bb * NH + h) * HD + d) * Ss + s)
          : (((size_t)(bb * NH + h) * Ss + s) * HD + d);
      Y[off] = (_Float16)val;
    }
  }
}

// ---------------------------------------------------------------------------
// Kernel 2: flash attention.  grid = (S/128, NH, B), block = 256 (8 waves).
// All 8 waves share one (b,h): K/V tiles for 128 keys are DMA'd into LDS by
// the Tensor Data Mover once per stage (wave 0 issues, TENSORcnt + barrier),
// then each wave runs online softmax for its 16-query tile.
// Q16,K16: [B,NH,S,HD] f16.  Vt16: [B,NH,HD,S] f16.  ctx16: [B,S,HID] f16.
// ---------------------------------------------------------------------------
__global__ __launch_bounds__(256)
void attn_kernel(const _Float16* __restrict__ Q16,
                 const _Float16* __restrict__ K16,
                 const _Float16* __restrict__ Vt16,
                 const float* __restrict__ mask,   // [B,1,1,S] -> [B*S]
                 _Float16* __restrict__ ctx16) {
  __shared__ _Float16 ldsK[128 * HD];     // 16 KB: [key 0..127][d 0..63]
  __shared__ _Float16 ldsV[HD * 128];     // 16 KB: [d 0..63][key 0..127]
  __shared__ _Float16 ldsP[8][16 * 32];   // 8 KB: per-wave P-transpose staging

  const int b = blockIdx.z, h = blockIdx.y;
  const int wave = threadIdx.x >> 5;
  const int lane = threadIdx.x & 31;
  const int nl = lane & 15, g = lane >> 4;
  const int q0 = blockIdx.x * 128 + wave * 16;

  const _Float16* Qh = Q16 + (size_t)(b * NH + h) * Ss * HD;
  const _Float16* Kh = K16 + (size_t)(b * NH + h) * Ss * HD;
  const _Float16* Vt = Vt16 + (size_t)(b * NH + h) * HD * Ss;
  const float* mrow_g = mask + (size_t)b * Ss;
  _Float16* Pbuf = &ldsP[wave][0];

  // Q fragments stay in registers for the whole key loop (2 k-chunks of 32)
  v16h qa0 = frag_a(Qh + (size_t)q0 * HD + 0, HD);
  v16h qa1 = frag_a(Qh + (size_t)q0 * HD + 32, HD);

  float mx_r[8], l_r[8];
#pragma unroll
  for (int r = 0; r < 8; ++r) { mx_r[r] = -1e30f; l_r[r] = 0.f; }
  v8f acc[4] = {};

  const float scale = 0.125f;   // 1/sqrt(64)

  for (int kb = 0; kb < Ss; kb += 128) {
    __syncthreads();            // previous stage fully consumed
    if (wave == 0) {
      // K tile: 128 rows x 64 d (contiguous rows of 64)
      tdm_load_2d_f16((unsigned int)(uintptr_t)ldsK,
                      Kh + (size_t)kb * HD,
                      /*tensor_d0=*/HD, /*tensor_d1=*/128,
                      /*tile_d0=*/HD, /*tile_d1=*/128, /*stride0=*/HD);
      // V tile: 64 rows x 128 keys (row stride S in the transposed tensor)
      tdm_load_2d_f16((unsigned int)(uintptr_t)ldsV,
                      Vt + kb,
                      /*tensor_d0=*/Ss - kb, /*tensor_d1=*/HD,
                      /*tile_d0=*/128, /*tile_d1=*/HD, /*stride0=*/Ss);
      __builtin_amdgcn_s_wait_tensorcnt(0);
    }
    __syncthreads();            // tiles ready for all 8 waves

    for (int kk = 0; kk < 128; kk += 32) {
      // ---- scores: two 16-key column tiles, contraction over HD=64 ----
      v8f s0 = {}, s1 = {};
      s0 = wmma32(qa0, frag_b(ldsK + (size_t)(kk)      * HD + 0,  HD), s0);
      s0 = wmma32(qa1, frag_b(ldsK + (size_t)(kk)      * HD + 32, HD), s0);
      s1 = wmma32(qa0, frag_b(ldsK + (size_t)(kk + 16) * HD + 0,  HD), s1);
      s1 = wmma32(qa1, frag_b(ldsK + (size_t)(kk + 16) * HD + 32, HD), s1);

      const float msk0 = mrow_g[kb + kk + nl];
      const float msk1 = mrow_g[kb + kk + 16 + nl];

      float mnew[8], alpha[8];
#pragma unroll
      for (int r = 0; r < 8; ++r) {
        s0[r] = s0[r] * scale + msk0;
        s1[r] = s1[r] * scale + msk1;
        float mx = fmaxf(s0[r], s1[r]);
#pragma unroll
        for (int off = 1; off < 16; off <<= 1)
          mx = fmaxf(mx, __shfl_xor(mx, off, 16));  // 16-lane row groups
        mnew[r] = fmaxf(mx_r[r], mx);
        alpha[r] = __expf(mx_r[r] - mnew[r]);
        mx_r[r] = mnew[r];
      }

#pragma unroll
      for (int r = 0; r < 8; ++r) {
        const float p0 = __expf(s0[r] - mnew[r]);
        const float p1 = __expf(s1[r] - mnew[r]);
        float ps = p0 + p1;
#pragma unroll
        for (int off = 1; off < 16; off <<= 1)
          ps += __shfl_xor(ps, off, 16);
        l_r[r] = l_r[r] * alpha[r] + ps;
        // C-layout (lane=col) -> row-major f16 P for A-fragment reload
        const int row = r + 8 * g;
        Pbuf[row * 32 + nl]      = (_Float16)p0;
        Pbuf[row * 32 + 16 + nl] = (_Float16)p1;
#pragma unroll
        for (int t = 0; t < 4; ++t) acc[t][r] *= alpha[r];
      }

      // same-wave DS store -> load ordering (wave-private Pbuf)
      asm volatile("s_wait_dscnt 0" ::: "memory");
      v16h pa = frag_a(Pbuf, 32);

      // ---- ctx += P(16x32) @ V(32x64): 4 output d-tiles ----
#pragma unroll
      for (int t = 0; t < 4; ++t) {
        v16h vb = frag_b(ldsV + (size_t)(t * 16) * 128 + kk, 128);
        acc[t] = wmma32(pa, vb, acc[t]);
      }
    }
  }

  // ---- epilogue: normalize by row sums, store ctx in [B,S,HID] f16 ----
#pragma unroll
  for (int t = 0; t < 4; ++t) {
#pragma unroll
    for (int r = 0; r < 8; ++r) {
      const int s_row = q0 + r + 8 * g;
      const int col = h * HD + t * 16 + nl;
      ctx16[((size_t)b * Ss + s_row) * HID + col] =
          (_Float16)(acc[t][r] / l_r[r]);
    }
  }
}

// ---------------------------------------------------------------------------
// Kernel 3: output projection + residual + LayerNorm (fused).
// grid.x = M/16 = 1024, block = 256.  h = ctx @ Wo^T + bo + resid; LN(h).
// ---------------------------------------------------------------------------
__global__ __launch_bounds__(256)
void outproj_ln_kernel(const _Float16* __restrict__ ctx16,
                       const _Float16* __restrict__ Wo16,
                       const float* __restrict__ bo,
                       const float* __restrict__ resid,
                       const float* __restrict__ ln_g,
                       const float* __restrict__ ln_b,
                       float* __restrict__ out) {
  __shared__ float hbuf[16 * 512];   // 32 KB

  const int m0 = blockIdx.x * 16;
  const int wave = threadIdx.x >> 5;
  const int lane = threadIdx.x & 31;
  const int nl = lane & 15, g = lane >> 4;
  const int n_base = wave * 64;

  v8f acc[4] = {};
  for (int k0 = 0; k0 < HID; k0 += 32) {
    v16h a = frag_a(ctx16 + (size_t)m0 * HID + k0, HID);
#pragma unroll
    for (int t = 0; t < 4; ++t) {
      v16h b = frag_b(Wo16 + (size_t)(n_base + t * 16) * HID + k0, HID);
      acc[t] = wmma32(a, b, acc[t]);
    }
  }

#pragma unroll
  for (int t = 0; t < 4; ++t) {
    const int n = n_base + t * 16 + nl;
    const float bval = bo[n];
#pragma unroll
    for (int r = 0; r < 8; ++r) {
      const int m = r + 8 * g;
      hbuf[m * 512 + n] =
          acc[t][r] + bval + resid[(size_t)(m0 + m) * HID + n];
    }
  }
  __syncthreads();

  // LayerNorm: 8 waves x 2 rows each; 512 elems per row, 16 per lane
#pragma unroll
  for (int rr = 0; rr < 2; ++rr) {
    const int m = wave * 2 + rr;
    float sum = 0.f, sq = 0.f;
    for (int c = lane; c < HID; c += 32) {
      const float v = hbuf[m * 512 + c];
      sum += v; sq += v * v;
    }
#pragma unroll
    for (int off = 1; off < 32; off <<= 1) {
      sum += __shfl_xor(sum, off, 32);
      sq  += __shfl_xor(sq,  off, 32);
    }
    const float mu  = sum * (1.f / 512.f);
    const float var = sq * (1.f / 512.f) - mu * mu;
    const float inv = rsqrtf(var + 1e-12f);
    for (int c = lane; c < HID; c += 32) {
      out[(size_t)(m0 + m) * HID + c] =
          (hbuf[m * 512 + c] - mu) * inv * ln_g[c] + ln_b[c];
    }
  }
}

// ---------------------------------------------------------------------------
// Host launch
// ---------------------------------------------------------------------------
extern "C" void kernel_launch(void* const* d_in, const int* in_sizes, int n_in,
                              void* d_out, int out_size, void* d_ws,
                              size_t ws_size, hipStream_t stream) {
  (void)in_sizes; (void)n_in; (void)out_size; (void)ws_size;

  const float* q_in    = (const float*)d_in[0];
  const float* k_in    = (const float*)d_in[1];
  const float* v_in    = (const float*)d_in[2];
  const float* layerin = (const float*)d_in[3];
  const float* mask    = (const float*)d_in[4];
  const float* Wq = (const float*)d_in[5];  const float* bq = (const float*)d_in[6];
  const float* Wk = (const float*)d_in[7];  const float* bk = (const float*)d_in[8];
  const float* Wv = (const float*)d_in[9];  const float* bv = (const float*)d_in[10];
  const float* Wo = (const float*)d_in[11]; const float* bo = (const float*)d_in[12];
  const float* ln_g = (const float*)d_in[13];
  const float* ln_b = (const float*)d_in[14];
  float* out = (float*)d_out;

  // workspace carve-out (256B aligned)
  char* ws = (char*)d_ws;
  size_t off = 0;
  auto carve = [&](size_t bytes) {
    void* p = ws + off;
    off += (bytes + 255) & ~(size_t)255;
    return p;
  };
  const size_t wbytes = (size_t)HID * HID * sizeof(_Float16);         // 512 KB
  const size_t tbytes = (size_t)Bb * NH * Ss * HD * sizeof(_Float16); // 16 MB
  _Float16* Wq16 = (_Float16*)carve(wbytes);
  _Float16* Wk16 = (_Float16*)carve(wbytes);
  _Float16* Wv16 = (_Float16*)carve(wbytes);
  _Float16* Wo16 = (_Float16*)carve(wbytes);
  _Float16* Q16  = (_Float16*)carve(tbytes);
  _Float16* K16  = (_Float16*)carve(tbytes);
  _Float16* Vt16 = (_Float16*)carve(tbytes);
  _Float16* ctx16 = (_Float16*)carve((size_t)Bb * Ss * HID * sizeof(_Float16));

  const int wn = HID * HID;
  const int cvtBlocks = (wn + 255) / 256;
  cvt_f32_f16<<<cvtBlocks, 256, 0, stream>>>(Wq, Wq16, wn);
  cvt_f32_f16<<<cvtBlocks, 256, 0, stream>>>(Wk, Wk16, wn);
  cvt_f32_f16<<<cvtBlocks, 256, 0, stream>>>(Wv, Wv16, wn);
  cvt_f32_f16<<<cvtBlocks, 256, 0, stream>>>(Wo, Wo16, wn);

  const int mTiles = (Bb * Ss) / 16;        // 1024
  proj_kernel<<<mTiles, 256, 0, stream>>>(q_in, Wq16, bq, Q16, 0);
  proj_kernel<<<mTiles, 256, 0, stream>>>(k_in, Wk16, bk, K16, 0);
  proj_kernel<<<mTiles, 256, 0, stream>>>(v_in, Wv16, bv, Vt16, 1);

  dim3 agrid(Ss / 128, NH, Bb);             // (16, 8, 8)
  attn_kernel<<<agrid, 256, 0, stream>>>(Q16, K16, Vt16, mask, ctx16);

  outproj_ln_kernel<<<mTiles, 256, 0, stream>>>(ctx16, Wo16, bo, layerin,
                                                ln_g, ln_b, out);
}
